// Transformer_88021059764555
// MI455X (gfx1250) — matrix operand order repaired and checked by reading, not compiled, not run on hardware
//
#include <hip/hip_runtime.h>
#include <hip/hip_bf16.h>

typedef _Float16 half_t;
typedef __attribute__((ext_vector_type(16))) _Float16 v16h;
typedef __attribute__((ext_vector_type(8)))  _Float16 v8h;
typedef __attribute__((ext_vector_type(8)))  float    v8f;

#define TB   4
#define TT   2048
#define TD   512
#define TN   8
#define TH   64
#define TF   2048
#define TM   (TB * TT)
#define CAPV 50.0f

// ---------------------------------------------------------------------------
// CDNA5 primitives
// ---------------------------------------------------------------------------
__device__ __forceinline__ v8f wmma32(v16h a, v16h b, v8f c) {
  return __builtin_amdgcn_wmma_f32_16x16x32_f16(false, a, false, b, (short)0, c,
                                                false, false);
}

// Async global->LDS copy, 16B per lane, tracked by ASYNCcnt.
__device__ __forceinline__ void async_b128(unsigned lds_off, const half_t* g) {
  unsigned long long ga = (unsigned long long)(uintptr_t)g;
  asm volatile("global_load_async_to_lds_b128 %0, %1, off"
               :
               : "v"(lds_off), "v"(ga)
               : "memory");
}
__device__ __forceinline__ void wait_async0() {
  asm volatile("s_wait_asynccnt 0" ::: "memory");
}
__device__ __forceinline__ unsigned lds_addr(const half_t* p) {
  return (unsigned)(uintptr_t)p;  // low 32 bits of generic LDS ptr = LDS offset
}

// 16x32 f16 fragment from LDS tile stored [row][stride] halves (ISA layout:
// lane<16: row=row0+lane, K={kb..kb+7, kb+16..23}; lane>=16: +8 K-offset).
__device__ __forceinline__ v16h lds_frag(const half_t* tile, int row0, int kb,
                                         int stride, int lane) {
  const int r = row0 + (lane & 15);
  const int koff = kb + ((lane >> 4) << 3);
  const half_t* p = tile + r * stride + koff;
  v16h f;
#pragma unroll
  for (int i = 0; i < 8; ++i) f[i] = p[i];
#pragma unroll
  for (int i = 0; i < 8; ++i) f[8 + i] = p[16 + i];
  return f;
}

// ---------------------------------------------------------------------------
// LayerNorm over D=512 -> f16 (one row per block)
// ---------------------------------------------------------------------------
__global__ __launch_bounds__(256) void tfb_layernorm_h(
    const float* __restrict__ x, const float* __restrict__ g,
    const float* __restrict__ b, half_t* __restrict__ out) {
  const int row = blockIdx.x;
  const int tid = threadIdx.x;
  const float* xr = x + (size_t)row * TD;
  float a0 = xr[tid], a1 = xr[tid + 256];
  __shared__ float red[256];
  red[tid] = a0 + a1;
  __syncthreads();
#pragma unroll
  for (int s = 128; s > 0; s >>= 1) {
    if (tid < s) red[tid] += red[tid + s];
    __syncthreads();
  }
  const float mu = red[0] * (1.0f / TD);
  __syncthreads();
  const float d0 = a0 - mu, d1 = a1 - mu;
  red[tid] = d0 * d0 + d1 * d1;
  __syncthreads();
#pragma unroll
  for (int s = 128; s > 0; s >>= 1) {
    if (tid < s) red[tid] += red[tid + s];
    __syncthreads();
  }
  const float inv = rsqrtf(red[0] * (1.0f / TD) + 1e-5f);
  half_t* o = out + (size_t)row * TD;
  o[tid]       = (half_t)(d0 * inv * g[tid] + b[tid]);
  o[tid + 256] = (half_t)(d1 * inv * g[tid + 256] + b[tid + 256]);
}

// ---------------------------------------------------------------------------
// f32 -> f16 plain convert (wo only)
// ---------------------------------------------------------------------------
__global__ __launch_bounds__(256) void tfb_cvt_h(const float* __restrict__ s,
                                                 half_t* __restrict__ d, int n) {
  int i = blockIdx.x * 256 + threadIdx.x;
  if (i < n) d[i] = (half_t)s[i];
}

// f32 (R x C) -> f16 transposed (C x R): dst[c*R + r] = src[r*C + c]
__global__ __launch_bounds__(256) void tfb_cvt_t_h(const float* __restrict__ src,
                                                   half_t* __restrict__ dst,
                                                   int R, int C) {
  __shared__ half_t t[32][40];
  const int tid = threadIdx.x;
  const int tx = tid & 31, ty = tid >> 5;
  const int r0 = blockIdx.y * 32, c0 = blockIdx.x * 32;
#pragma unroll
  for (int i = 0; i < 32; i += 8)
    t[ty + i][tx] = (half_t)src[(size_t)(r0 + ty + i) * C + c0 + tx];
  __syncthreads();
#pragma unroll
  for (int i = 0; i < 32; i += 8)
    dst[(size_t)(c0 + ty + i) * R + r0 + tx] = t[tx][ty + i];
}

// f16 (R x C) -> f16 transposed (C x R) per blockIdx.z matrix
__global__ __launch_bounds__(256) void tfb_transpose_h(
    const half_t* __restrict__ src, half_t* __restrict__ dst, int R, int C) {
  const half_t* s = src + (size_t)blockIdx.z * R * C;
  half_t* d = dst + (size_t)blockIdx.z * R * C;
  __shared__ half_t t[32][40];
  const int tid = threadIdx.x;
  const int tx = tid & 31, ty = tid >> 5;
  const int r0 = blockIdx.y * 32, c0 = blockIdx.x * 32;
#pragma unroll
  for (int i = 0; i < 32; i += 8)
    t[ty + i][tx] = s[(size_t)(r0 + ty + i) * C + c0 + tx];
  __syncthreads();
#pragma unroll
  for (int i = 0; i < 32; i += 8)
    d[(size_t)(c0 + ty + i) * R + r0 + tx] = t[tx][ty + i];
}

// ---------------------------------------------------------------------------
// WMMA GEMM: C[M,N] = act((A[M,K] * Bt[N,K]^T + bias) * scale) (+res)
// A f16 row-major (lda=K); Bt f16 stored [n][k] (row-major over k).
// Block 64(m) x 128(n), BK=32, 8 waves (2x4) of 32x32.
// Double-buffered LDS, filled with global_load_async_to_lds_b128.
// ---------------------------------------------------------------------------
#define LK 40  // padded LDS row stride (halves): conflict-free b128 access
__global__ __launch_bounds__(256) void tfb_wmma_gemm(
    const half_t* __restrict__ A, const half_t* __restrict__ Bt,
    const float* __restrict__ bias, const float* __restrict__ residual,
    float* __restrict__ outF, half_t* __restrict__ outH, int M, int N, int K,
    float scale, int relu) {
  const int tid = threadIdx.x, lane = tid & 31, wave = tid >> 5;
  const int wm = wave >> 2, wn = wave & 3;
  const int m0 = blockIdx.x * 64, n0 = blockIdx.y * 128;

  __shared__ half_t As[2][64 * LK];
  __shared__ half_t Bs[2][128 * LK];

  const int ar = tid >> 2, ac = (tid & 3) * 8;   // A: 16B / thread
  const int br = tid >> 1, bc = (tid & 1) * 16;  // B: 32B / thread
  const half_t* aS = A + (size_t)(m0 + ar) * K + ac;
  const half_t* bS = Bt + (size_t)(n0 + br) * K + bc;
  unsigned aD[2] = {lds_addr(&As[0][ar * LK + ac]),
                    lds_addr(&As[1][ar * LK + ac])};
  unsigned bD[2] = {lds_addr(&Bs[0][br * LK + bc]),
                    lds_addr(&Bs[1][br * LK + bc])};

  v8f acc[2][2] = {};

  // prologue: fill buffer 0
  async_b128(aD[0], aS);
  async_b128(bD[0], bS);
  async_b128(bD[0] + 16, bS + 8);

  const int KT = K >> 5;
  for (int kt = 0; kt < KT; ++kt) {
    const int cur = kt & 1;
    wait_async0();
    __syncthreads();
    if (kt + 1 < KT) {  // overlap next tile's async fill with this tile's math
      const int nk = (kt + 1) << 5;
      async_b128(aD[cur ^ 1], aS + nk);
      async_b128(bD[cur ^ 1], bS + nk);
      async_b128(bD[cur ^ 1] + 16, bS + nk + 8);
    }
    const half_t* At = As[cur];
    const half_t* Bp = Bs[cur];
    const v16h a0 = lds_frag(At, wm * 32, 0, LK, lane);
    const v16h a1 = lds_frag(At, wm * 32 + 16, 0, LK, lane);
    const v16h b0 = lds_frag(Bp, wn * 32, 0, LK, lane);
    const v16h b1 = lds_frag(Bp, wn * 32 + 16, 0, LK, lane);
    acc[0][0] = wmma32(a0, b0, acc[0][0]);
    acc[0][1] = wmma32(a0, b1, acc[0][1]);
    acc[1][0] = wmma32(a1, b0, acc[1][0]);
    acc[1][1] = wmma32(a1, b1, acc[1][1]);
  }

  const int colb = lane & 15;
  const int rowb = (lane >> 4) * 8;
#pragma unroll
  for (int ti = 0; ti < 2; ++ti)
#pragma unroll
    for (int tj = 0; tj < 2; ++tj) {
      const int gn = n0 + wn * 32 + tj * 16 + colb;
      const float bv = bias ? bias[gn] : 0.0f;
#pragma unroll
      for (int i = 0; i < 8; ++i) {
        const int gm = m0 + wm * 32 + ti * 16 + rowb + i;
        float v = (acc[ti][tj][i] + bv) * scale;
        if (relu) v = fmaxf(v, 0.0f);
        if (residual) v += residual[(size_t)gm * N + gn];
        if (outF) outF[(size_t)gm * N + gn] = v;
        if (outH) outH[(size_t)gm * N + gn] = (half_t)v;
      }
    }
}

// ---------------------------------------------------------------------------
// Attention logits: S = cap*tanh(qk/cap) (q pre-scaled). 64(t)x128(s) tile,
// K=H=64 fully staged via async loads.
// ---------------------------------------------------------------------------
#define LK2 72
__global__ __launch_bounds__(256) void tfb_attn_logits(
    const half_t* __restrict__ qh, const half_t* __restrict__ kh,
    float* __restrict__ probs) {
  const int bn = blockIdx.z, b = bn >> 3, n = bn & 7;
  const int tid = threadIdx.x, lane = tid & 31, wave = tid >> 5;
  const int wm = wave >> 2, wn = wave & 3;
  const int t0 = blockIdx.x * 64, s0 = blockIdx.y * 128;
  const half_t* Q = qh + (size_t)b * TT * (TN * TH) + n * TH;
  const half_t* Kp = kh + (size_t)b * TT * (TN * TH) + n * TH;
  float* S = probs + (size_t)bn * TT * TT;

  __shared__ half_t As[64 * LK2];
  __shared__ half_t Bs[128 * LK2];

  {  // Q tile: 64 x 64 halves, 32B / thread -> 2 async b128
    const int r = tid >> 2, c = (tid & 3) * 16;
    const half_t* src = Q + (size_t)(t0 + r) * (TN * TH) + c;
    const unsigned d = lds_addr(&As[r * LK2 + c]);
    async_b128(d, src);
    async_b128(d + 16, src + 8);
  }
  {  // K tile: 128 x 64 halves, 64B / thread -> 4 async b128
    const int r = tid >> 1, c = (tid & 1) * 32;
    const half_t* src = Kp + (size_t)(s0 + r) * (TN * TH) + c;
    const unsigned d = lds_addr(&Bs[r * LK2 + c]);
    async_b128(d, src);
    async_b128(d + 16, src + 8);
    async_b128(d + 32, src + 16);
    async_b128(d + 48, src + 24);
  }
  wait_async0();
  __syncthreads();

  v8f acc[2][2] = {};
#pragma unroll
  for (int kk = 0; kk < 64; kk += 32) {
    const v16h a0 = lds_frag(As, wm * 32, kk, LK2, lane);
    const v16h a1 = lds_frag(As, wm * 32 + 16, kk, LK2, lane);
    const v16h b0 = lds_frag(Bs, wn * 32, kk, LK2, lane);
    const v16h b1 = lds_frag(Bs, wn * 32 + 16, kk, LK2, lane);
    acc[0][0] = wmma32(a0, b0, acc[0][0]);
    acc[0][1] = wmma32(a0, b1, acc[0][1]);
    acc[1][0] = wmma32(a1, b0, acc[1][0]);
    acc[1][1] = wmma32(a1, b1, acc[1][1]);
  }

  const int colb = lane & 15;
  const int rowb = (lane >> 4) * 8;
#pragma unroll
  for (int ti = 0; ti < 2; ++ti)
#pragma unroll
    for (int tj = 0; tj < 2; ++tj) {
      const int gs = s0 + wn * 32 + tj * 16 + colb;
#pragma unroll
      for (int i = 0; i < 8; ++i) {
        const int gt = t0 + wm * 32 + ti * 16 + rowb + i;
        S[(size_t)gt * TT + gs] = CAPV * tanhf(acc[ti][tj][i] * (1.0f / CAPV));
      }
    }
}

// ---------------------------------------------------------------------------
// Row softmax in-place (row length T = 2048, one block per row)
// ---------------------------------------------------------------------------
__global__ __launch_bounds__(256) void tfb_softmax_rows(float* __restrict__ p) {
  float* r = p + (size_t)blockIdx.x * TT;
  const int tid = threadIdx.x;
  float v[8];
  float mx = -3.0e38f;
#pragma unroll
  for (int i = 0; i < 8; ++i) {
    v[i] = r[tid + i * 256];
    mx = fmaxf(mx, v[i]);
  }
  __shared__ float red[256];
  red[tid] = mx;
  __syncthreads();
#pragma unroll
  for (int s = 128; s > 0; s >>= 1) {
    if (tid < s) red[tid] = fmaxf(red[tid], red[tid + s]);
    __syncthreads();
  }
  mx = red[0];
  __syncthreads();
  float sum = 0.0f;
#pragma unroll
  for (int i = 0; i < 8; ++i) {
    v[i] = __expf(v[i] - mx);
    sum += v[i];
  }
  red[tid] = sum;
  __syncthreads();
#pragma unroll
  for (int s = 128; s > 0; s >>= 1) {
    if (tid < s) red[tid] += red[tid + s];
    __syncthreads();
  }
  const float inv = 1.0f / red[0];
#pragma unroll
  for (int i = 0; i < 8; ++i) r[tid + i * 256] = v[i] * inv;
}

// ---------------------------------------------------------------------------
// enc = probs @ V per head. Block 64(t) x 64(h), 4 waves (2x2), BK=32.
// V pre-transposed per head: vt[bn][h][s]. Double-buffered: V via async,
// probs converted f32->f16 while staging.
// ---------------------------------------------------------------------------
__global__ __launch_bounds__(128) void tfb_attn_pv(
    const float* __restrict__ probs, const half_t* __restrict__ vt,
    half_t* __restrict__ ench) {
  const int bn = blockIdx.z, b = bn >> 3, n = bn & 7;
  const int tid = threadIdx.x, lane = tid & 31, wave = tid >> 5;
  const int wm = wave >> 1, wn = wave & 1;
  const int t0 = blockIdx.x * 64;
  const float* P = probs + (size_t)bn * TT * TT;
  const half_t* Vt = vt + (size_t)bn * TH * TT;  // [h][s]

  __shared__ half_t As[2][64 * LK];
  __shared__ half_t Bs[2][64 * LK];

  const int ar = tid >> 1, ac = (tid & 1) * 16;  // A: 16 f32 -> 16 f16
  const int brr = tid >> 1, bcc = (tid & 1) * 16;  // B: 32B / thread
  const half_t* bS = Vt + (size_t)brr * TT + bcc;
  unsigned bD[2] = {lds_addr(&Bs[0][brr * LK + bcc]),
                    lds_addr(&Bs[1][brr * LK + bcc])};

  v8f acc[2][2] = {};

  // prologue buffer 0
  {
    const float* src = P + (size_t)(t0 + ar) * TT + ac;
    half_t* dst = &As[0][ar * LK + ac];
#pragma unroll
    for (int i = 0; i < 16; ++i) dst[i] = (half_t)src[i];
    async_b128(bD[0], bS);
    async_b128(bD[0] + 16, bS + 8);
  }

  for (int kt = 0; kt < (TT >> 5); ++kt) {
    const int cur = kt & 1;
    wait_async0();
    __syncthreads();
    if (kt + 1 < (TT >> 5)) {
      const int nk = (kt + 1) << 5;
      async_b128(bD[cur ^ 1], bS + nk);
      async_b128(bD[cur ^ 1] + 16, bS + nk + 8);
      const float* src = P + (size_t)(t0 + ar) * TT + nk + ac;
      half_t* dst = &As[cur ^ 1][ar * LK + ac];
#pragma unroll
      for (int i = 0; i < 16; ++i) dst[i] = (half_t)src[i];
    }
    const half_t* At = As[cur];
    const half_t* Bp = Bs[cur];
    const v16h a0 = lds_frag(At, wm * 32, 0, LK, lane);
    const v16h a1 = lds_frag(At, wm * 32 + 16, 0, LK, lane);
    const v16h b0 = lds_frag(Bp, wn * 32, 0, LK, lane);
    const v16h b1 = lds_frag(Bp, wn * 32 + 16, 0, LK, lane);
    acc[0][0] = wmma32(a0, b0, acc[0][0]);
    acc[0][1] = wmma32(a0, b1, acc[0][1]);
    acc[1][0] = wmma32(a1, b0, acc[1][0]);
    acc[1][1] = wmma32(a1, b1, acc[1][1]);
  }

  const int colb = lane & 15;
  const int rowb = (lane >> 4) * 8;
#pragma unroll
  for (int ti = 0; ti < 2; ++ti)
#pragma unroll
    for (int tj = 0; tj < 2; ++tj) {
      const int gh = wn * 32 + tj * 16 + colb;
#pragma unroll
      for (int i = 0; i < 8; ++i) {
        const int gt = t0 + wm * 32 + ti * 16 + rowb + i;
        ench[(size_t)(b * TT + gt) * (TN * TH) + n * TH + gh] =
            (half_t)acc[ti][tj][i];
      }
    }
}

// ---------------------------------------------------------------------------
// Host launcher
// ---------------------------------------------------------------------------
extern "C" void kernel_launch(void* const* d_in, const int* in_sizes, int n_in,
                              void* d_out, int out_size, void* d_ws,
                              size_t ws_size, hipStream_t stream) {
  const float* x     = (const float*)d_in[0];
  const float* ln1_g = (const float*)d_in[1];
  const float* ln1_b = (const float*)d_in[2];
  const float* wq    = (const float*)d_in[3];
  const float* bq    = (const float*)d_in[4];
  const float* wk    = (const float*)d_in[5];
  const float* bk    = (const float*)d_in[6];
  const float* wv    = (const float*)d_in[7];
  const float* bv    = (const float*)d_in[8];
  const float* wo    = (const float*)d_in[9];
  const float* bo    = (const float*)d_in[10];
  const float* ln2_g = (const float*)d_in[11];
  const float* ln2_b = (const float*)d_in[12];
  const float* w1    = (const float*)d_in[13];
  const float* b1    = (const float*)d_in[14];
  const float* w2    = (const float*)d_in[15];
  const float* b2    = (const float*)d_in[16];

  float* out0  = (float*)d_out;                // (B,T,D)
  float* probs = out0 + (size_t)TB * TT * TD;  // (B,N,T,T)

  char* ws = (char*)d_ws;
  size_t off = 0;
  auto carve = [&](size_t bytes) {
    void* p = ws + off;
    off += (bytes + 255) & ~(size_t)255;
    return p;
  };
  half_t* xh       = (half_t*)carve((size_t)TM * TD * 2);
  half_t* wq_t     = (half_t*)carve((size_t)TD * TD * 2);   // [nh][d]
  half_t* wk_t     = (half_t*)carve((size_t)TD * TD * 2);
  half_t* wv_t     = (half_t*)carve((size_t)TD * TD * 2);
  half_t* wo_h     = (half_t*)carve((size_t)TD * TD * 2);   // [d][nh] (as-is)
  half_t* w1_t     = (half_t*)carve((size_t)TD * TF * 2);   // [f][d]
  half_t* w2_t     = (half_t*)carve((size_t)TF * TD * 2);   // [d][f]
  half_t* q_h      = (half_t*)carve((size_t)TM * TD * 2);
  half_t* k_h      = (half_t*)carve((size_t)TM * TD * 2);
  half_t* v_h      = (half_t*)carve((size_t)TM * TD * 2);
  half_t* vt       = (half_t*)carve((size_t)TM * TD * 2);   // [bn][h][s]
  half_t* enc_h    = (half_t*)carve((size_t)TM * TD * 2);
  float*  attn_out = (float*)carve((size_t)TM * TD * 4);
  half_t* x2h      = (half_t*)carve((size_t)TM * TD * 2);
  half_t* h1_h     = (half_t*)carve((size_t)TM * TF * 2);

  // 1. LN1 -> f16
  tfb_layernorm_h<<<TM, 256, 0, stream>>>(x, ln1_g, ln1_b, xh);

  // 2. weight convert (+transpose to [n][k] so all GEMM B-loads are async b128)
  tfb_cvt_t_h<<<dim3(TD / 32, TD / 32), 256, 0, stream>>>(wq, wq_t, TD, TD);
  tfb_cvt_t_h<<<dim3(TD / 32, TD / 32), 256, 0, stream>>>(wk, wk_t, TD, TD);
  tfb_cvt_t_h<<<dim3(TD / 32, TD / 32), 256, 0, stream>>>(wv, wv_t, TD, TD);
  tfb_cvt_h<<<(TD * TD) / 256, 256, 0, stream>>>(wo, wo_h, TD * TD);
  tfb_cvt_t_h<<<dim3(TF / 32, TD / 32), 256, 0, stream>>>(w1, w1_t, TD, TF);
  tfb_cvt_t_h<<<dim3(TD / 32, TF / 32), 256, 0, stream>>>(w2, w2_t, TF, TD);

  // 3. QKV projections (q pre-scaled by H^-1/2)
  dim3 gqkv(TM / 64, TD / 128);
  tfb_wmma_gemm<<<gqkv, 256, 0, stream>>>(xh, wq_t, bq, nullptr, nullptr, q_h,
                                          TM, TD, TD, 0.125f, 0);
  tfb_wmma_gemm<<<gqkv, 256, 0, stream>>>(xh, wk_t, bk, nullptr, nullptr, k_h,
                                          TM, TD, TD, 1.0f, 0);
  tfb_wmma_gemm<<<gqkv, 256, 0, stream>>>(xh, wv_t, bv, nullptr, nullptr, v_h,
                                          TM, TD, TD, 1.0f, 0);

  // 3b. V -> per-head transposed [bn][h][s]
  tfb_transpose_h<<<dim3(TD / 32, TT / 32, TB), 256, 0, stream>>>(v_h, vt, TT,
                                                                  TD);

  // 4. logits + soft-cap -> probs region of d_out
  dim3 gl(TT / 64, TT / 128, TB * TN);
  tfb_attn_logits<<<gl, 256, 0, stream>>>(q_h, k_h, probs);

  // 5. softmax in place
  tfb_softmax_rows<<<TB * TN * TT, 256, 0, stream>>>(probs);

  // 6. enc = probs @ V
  dim3 gpv(TT / 64, 1, TB * TN);
  tfb_attn_pv<<<gpv, 128, 0, stream>>>(probs, vt, enc_h);

  // 7. output projection + residual x -> attn_out (f32)
  tfb_wmma_gemm<<<gqkv, 256, 0, stream>>>(enc_h, wo_h, bo, x, attn_out, nullptr,
                                          TM, TD, TD, 1.0f, 0);

  // 8. LN2 -> f16
  tfb_layernorm_h<<<TM, 256, 0, stream>>>(attn_out, ln2_g, ln2_b, x2h);

  // 9. FFN1 (relu) -> f16
  dim3 gf1(TM / 64, TF / 128);
  tfb_wmma_gemm<<<gf1, 256, 0, stream>>>(x2h, w1_t, b1, nullptr, nullptr, h1_h,
                                         TM, TF, TD, 1.0f, 1);

  // 10. FFN2 + residual attn_out -> d_out
  tfb_wmma_gemm<<<gqkv, 256, 0, stream>>>(h1_h, w2_t, b2, attn_out, out0,
                                          nullptr, TM, TD, TF, 1.0f, 0);

  (void)in_sizes; (void)n_in; (void)out_size; (void)ws_size;
}
